// Block_369367187931
// MI455X (gfx1250) — compile-verified, matrix operand order checked by reading, not run on hardware
//
#include <hip/hip_runtime.h>
#include <cstdint>
#include <cstddef>

// ---------------- problem constants ----------------
#define EMBED   768
#define HEADS   12
#define HDIM    64
#define MLP_H   3072
#define SEQ     1024
#define BATCH   8
#define TOKENS  (BATCH * SEQ)          // 8192
#define LN_EPS  1e-5f
#define ATT_SCALE 0.125f               // 1/sqrt(64)

typedef __bf16 bf16_t;
typedef __attribute__((ext_vector_type(16))) __bf16 v16bf;
typedef __attribute__((ext_vector_type(8)))  float  v8f;
typedef __attribute__((ext_vector_type(4)))  unsigned uint4v;
typedef __attribute__((ext_vector_type(8)))  int      int8v;
typedef __attribute__((ext_vector_type(4)))  int      int4v;

// ---------------- WMMA helper ----------------
__device__ __forceinline__ v8f wmma_bf16(v16bf a, v16bf b, v8f c) {
  return __builtin_amdgcn_wmma_f32_16x16x32_bf16(
      /*neg_a=*/false, a, /*neg_b=*/false, b,
      /*c_mod=*/(short)0, c, /*reuse_a=*/false, /*reuse_b=*/false);
}

// K-offset pattern for 16-bit A/B operands (ISA 7.12.2):
// VGPR j<4 -> K = 2j + 8*half ; j>=4 -> K = 16 + 2(j-4) + 8*half
__device__ __forceinline__ int kk_of(int j, int half) {
  return (j < 4 ? 2 * j : 16 + 2 * (j - 4)) + 8 * half;
}

// Load a 16x32 (rows x K) operand fragment; K pairs contiguous in memory.
// lane%16 selects row (A: M row / B: N column), lane/16 selects K half.
__device__ __forceinline__ v16bf load_frag(const bf16_t* p_base, int stride) {
  const int lane = threadIdx.x & 31;
  const int r = lane & 15, half = lane >> 4;
  const bf16_t* row = p_base + (size_t)r * stride;
  union { v16bf v; unsigned u[8]; } f;
#pragma unroll
  for (int j = 0; j < 8; ++j)
    f.u[j] = *(const unsigned*)(row + kk_of(j, half));   // aligned: kk even
  return f.v;
}

// ---------------- Tensor Data Mover: 2D bf16 tile -> LDS ----------------
// D# per ISA 8.3/8.4. Tile rows are K-contiguous (dim0), row stride = stride0
// elements. pad_code: pad 1 dword into LDS every (8 << pad_code) bytes.
__device__ __forceinline__ void tdm_load_tile(unsigned lds_addr,
                                              const bf16_t* gptr,
                                              unsigned tensor_d0, unsigned tensor_d1,
                                              unsigned long long stride0_elems,
                                              unsigned tile_d0, unsigned tile_d1,
                                              unsigned pad_code) {
  const unsigned long long ga = (unsigned long long)gptr;
  uint4v g0;
  g0[0] = 1u;                                           // count=1, user D#
  g0[1] = lds_addr;                                     // bits 63:32 lds_addr
  g0[2] = (unsigned)(ga & 0xFFFFFFFFu);                 // global_addr[31:0]
  g0[3] = (unsigned)((ga >> 32) & 0x01FFFFFFu)          // global_addr[56:32]
        | (2u << 30);                                   // type = 2 (image)
  int8v g1;
  g1[0] = (int)((1u << 16)              // data_size = 1 -> 2 bytes
              | (1u << 20)              // pad_enable
              | (pad_code << 22)        // pad_interval
              | (0u << 25));            // pad_amount = 0 -> 1 dword
  g1[1] = (int)((tensor_d0 & 0xFFFFu) << 16);
  g1[2] = (int)((tensor_d0 >> 16) | ((tensor_d1 & 0xFFFFu) << 16));
  g1[3] = (int)((tensor_d1 >> 16) | (tile_d0 << 16));
  g1[4] = (int)(tile_d1 & 0xFFFFu);                     // tile_dim2 = 0
  g1[5] = (int)(stride0_elems & 0xFFFFFFFFull);
  g1[6] = (int)((stride0_elems >> 32) & 0xFFFFull);     // dim1_stride = 0
  g1[7] = 0;
  const int4v z4 = {0, 0, 0, 0};
#if defined(__clang_major__) && (__clang_major__ >= 23)
  const int8v z8 = {0, 0, 0, 0, 0, 0, 0, 0};
  __builtin_amdgcn_tensor_load_to_lds(g0, g1, z4, z4, z8, 0);
#else
  __builtin_amdgcn_tensor_load_to_lds(g0, g1, z4, z4, 0);
#endif
}

__device__ __forceinline__ unsigned lds_offset_of(const void* p) {
  // generic LDS pointer = {shared aperture, offset}; low 32 bits = LDS offset
  return (unsigned)(unsigned long long)p;
}

// ---------------- weight convert+transpose: fp32 W[K][N] -> bf16 WT[N][K] ----------------
__global__ __launch_bounds__(256)
void transpose_w(const float* __restrict__ w, bf16_t* __restrict__ wt, int K, int N) {
  __shared__ float t[32][33];
  const int k0 = blockIdx.x * 32, n0 = blockIdx.y * 32;
  const int c = threadIdx.x & 31, r0 = threadIdx.x >> 5;    // 8 row-strips
#pragma unroll
  for (int r = r0; r < 32; r += 8)
    t[r][c] = w[(size_t)(k0 + r) * N + n0 + c];
  __syncthreads();
#pragma unroll
  for (int r = r0; r < 32; r += 8)
    wt[(size_t)(n0 + r) * K + k0 + c] = (bf16_t)t[c][r];
}

// ---------------- LayerNorm: fp32 row -> bf16 normalized row ----------------
__global__ __launch_bounds__(256)
void ln_kernel(const float* __restrict__ x, const float* __restrict__ g,
               const float* __restrict__ bta, bf16_t* __restrict__ out) {
  const int row = blockIdx.x;
  const float* xr = x + (size_t)row * EMBED;
  __shared__ float r1[256], r2[256];
  float s = 0.f, s2 = 0.f;
  for (int i = threadIdx.x; i < EMBED; i += 256) {
    float v = xr[i];
    s += v; s2 += v * v;
  }
  r1[threadIdx.x] = s; r2[threadIdx.x] = s2;
  __syncthreads();
  for (int off = 128; off > 0; off >>= 1) {
    if (threadIdx.x < off) {
      r1[threadIdx.x] += r1[threadIdx.x + off];
      r2[threadIdx.x] += r2[threadIdx.x + off];
    }
    __syncthreads();
  }
  const float mu  = r1[0] * (1.0f / EMBED);
  const float var = r2[0] * (1.0f / EMBED) - mu * mu;
  const float rstd = rsqrtf(var + LN_EPS);
  bf16_t* orow = out + (size_t)row * EMBED;
  for (int i = threadIdx.x; i < EMBED; i += 256)
    orow[i] = (bf16_t)((xr[i] - mu) * rstd * g[i] + bta[i]);
}

// ---------------- bf16 WMMA GEMM, double-buffered TDM tiles ----------------
// C[M,N] = A[M,K](bf16 row-major) * Bt[N,K](bf16, pre-transposed)^T
// Block tile 128x64, K-step 64; ping-pong LDS fed by the Tensor Data Mover so
// the DMA of step s+1 overlaps the 8 WMMAs of step s (wait tensorcnt<=2).
enum { EPI_QK = 0, EPI_V = 1, EPI_PROJ = 2, EPI_GELU = 3, EPI_OUT = 4 };

template <int EPI>
__global__ __launch_bounds__(256)
void gemm_bf16(const bf16_t* __restrict__ A, const bf16_t* __restrict__ Bt,
               int M, int N, int K,
               const float* __restrict__ bias, const float* __restrict__ resid,
               float* __restrict__ outf, bf16_t* __restrict__ outb,
               bf16_t* __restrict__ outb2) {
  const int bm = blockIdx.x * 128;
  const int bn = blockIdx.y * 64;
  const int wave   = threadIdx.x >> 5;
  const int lane   = threadIdx.x & 31;
  const int wave_m = wave >> 1;          // 0..3 -> 32-row strip
  const int wave_n = wave & 1;           // 0..1 -> 32-col strip

  __shared__ bf16_t As[2][128][66];      // +1 dword TDM pad per 128B row
  __shared__ bf16_t Bs[2][64][66];

  v8f acc[2][2] = {{{}, {}}, {{}, {}}};
  const int nstep = K >> 6;              // K / 64

  // preload step 0 into buffer 0
  if (threadIdx.x < 32) {
    tdm_load_tile(lds_offset_of(&As[0][0][0]), A + (size_t)bm * K,
                  (unsigned)K, (unsigned)M, (unsigned long long)K, 64u, 128u, 4u);
    tdm_load_tile(lds_offset_of(&Bs[0][0][0]), Bt + (size_t)bn * K,
                  (unsigned)K, (unsigned)N, (unsigned long long)K, 64u, 64u, 4u);
  }

  for (int s = 0; s < nstep; ++s) {
    const int c = s & 1;
    if (threadIdx.x < 32) {              // wave 0 drives the Tensor Data Mover
      if (s + 1 < nstep) {
        const int k1 = (s + 1) << 6;
        tdm_load_tile(lds_offset_of(&As[1 - c][0][0]), A + (size_t)bm * K + k1,
                      (unsigned)K, (unsigned)M, (unsigned long long)K, 64u, 128u, 4u);
        tdm_load_tile(lds_offset_of(&Bs[1 - c][0][0]), Bt + (size_t)bn * K + k1,
                      (unsigned)K, (unsigned)N, (unsigned long long)K, 64u, 64u, 4u);
        __builtin_amdgcn_s_wait_tensorcnt(2);  // current buffer done; next in flight
      } else {
        __builtin_amdgcn_s_wait_tensorcnt(0);
      }
    }
    __syncthreads();                     // publish buffer c to all waves

#pragma unroll
    for (int kc = 0; kc < 2; ++kc) {
      v16bf a0 = load_frag(&As[c][wave_m * 32][kc * 32], 66);
      v16bf a1 = load_frag(&As[c][wave_m * 32 + 16][kc * 32], 66);
      v16bf b0 = load_frag(&Bs[c][wave_n * 32][kc * 32], 66);
      v16bf b1 = load_frag(&Bs[c][wave_n * 32 + 16][kc * 32], 66);
      acc[0][0] = wmma_bf16(a0, b0, acc[0][0]);
      acc[0][1] = wmma_bf16(a0, b1, acc[0][1]);
      acc[1][0] = wmma_bf16(a1, b0, acc[1][0]);
      acc[1][1] = wmma_bf16(a1, b1, acc[1][1]);
    }
    __syncthreads();                     // all waves done with buffer c
  }

  // ---- epilogue: C layout lane%16 = N col, VGPR r -> M = r + 8*(lane/16) ----
  const int n16 = lane & 15, half = lane >> 4;
#pragma unroll
  for (int mt = 0; mt < 2; ++mt)
#pragma unroll
    for (int nt = 0; nt < 2; ++nt)
#pragma unroll
      for (int r = 0; r < 8; ++r) {
        const int row = bm + wave_m * 32 + mt * 16 + r + 8 * half;
        const int col = bn + wave_n * 32 + nt * 16 + n16;
        float val = acc[mt][nt][r];
        if constexpr (EPI == EPI_QK) {
          const int b = row >> 10, nn = row & 1023;
          int cc = col;
          if (cc < EMBED) {                       // q: [B*H, N, 64]
            const int hh = cc >> 6, dd = cc & 63;
            outb[(((size_t)(b * HEADS + hh) << 10) + nn) * HDIM + dd] = (bf16_t)val;
          } else {                                // k: [B*H, N, 64]
            cc -= EMBED;
            const int hh = cc >> 6, dd = cc & 63;
            outb2[(((size_t)(b * HEADS + hh) << 10) + nn) * HDIM + dd] = (bf16_t)val;
          }
        } else if constexpr (EPI == EPI_V) {      // v TRANSPOSED: [B*H, 64, N]
          const int b = row >> 10, nn = row & 1023;
          const int hh = col >> 6, dd = col & 63;
          outb[((size_t)(b * HEADS + hh) * HDIM + dd) * SEQ + nn] = (bf16_t)val;
        } else if constexpr (EPI == EPI_PROJ) {
          outf[(size_t)row * N + col] = val + bias[col] + resid[(size_t)row * N + col];
        } else if constexpr (EPI == EPI_GELU) {
          val += bias[col];
          const float ge = 0.5f * val * (1.0f + erff(val * 0.70710678118654752f));
          outb[(size_t)row * N + col] = (bf16_t)ge;
        } else { // EPI_OUT
          outf[(size_t)row * N + col] = val + bias[col] + resid[(size_t)row * N + col];
        }
      }
}

// ---------------- Flash-style attention, one (b,h) per blockIdx.y ----------------
// q,k: [B*H, SEQ, 64] bf16 ; vT: [B*H, 64, SEQ] bf16 ; o: [B, SEQ, EMBED] bf16
__global__ __launch_bounds__(128)
void attn_kernel(const bf16_t* __restrict__ q, const bf16_t* __restrict__ k,
                 const bf16_t* __restrict__ vT, bf16_t* __restrict__ o) {
  const int bh = blockIdx.y;
  const int b = bh / HEADS, h = bh % HEADS;
  const int wave = threadIdx.x >> 5;
  const int lane = threadIdx.x & 31;
  const int n16 = lane & 15, half = lane >> 4;
  const int qrow0 = blockIdx.x * 64 + wave * 16;

  const bf16_t* qb  = q  + ((size_t)bh * SEQ + qrow0) * HDIM;
  const bf16_t* kb  = k  + (size_t)bh * SEQ * HDIM;
  const bf16_t* vTb = vT + (size_t)bh * HDIM * SEQ;

  // Q fragments (16 rows x 64 K) as two K=32 chunks; resident entire kernel.
  const v16bf aq0 = load_frag(qb, HDIM);
  const v16bf aq1 = load_frag(qb + 32, HDIM);

  v8f accO[4] = {{}, {}, {}, {}};
  float mrun[8], lrun[8];
#pragma unroll
  for (int r = 0; r < 8; ++r) { mrun[r] = -3.0e38f; lrun[r] = 0.f; }

  __shared__ bf16_t Plds[4][16][34];     // per-wave P staging (A layout source)
  bf16_t(*Pw)[34] = Plds[wave];

  for (int j0 = 0; j0 < SEQ; j0 += 32) {
    // ---- S = q . k^T for two 16-key subtiles ----
    v8f s0 = {}, s1 = {};
    {
      const bf16_t* kb0 = kb + (size_t)j0 * HDIM;
      s0 = wmma_bf16(aq0, load_frag(kb0, HDIM), s0);
      s0 = wmma_bf16(aq1, load_frag(kb0 + 32, HDIM), s0);
      const bf16_t* kb1 = kb + (size_t)(j0 + 16) * HDIM;
      s1 = wmma_bf16(aq0, load_frag(kb1, HDIM), s1);
      s1 = wmma_bf16(aq1, load_frag(kb1 + 32, HDIM), s1);
    }
    // ---- online softmax over this 32-key block ----
    float alpha[8];
#pragma unroll
    for (int r = 0; r < 8; ++r) {
      float p0 = (float)s0[r] * ATT_SCALE;
      float p1 = (float)s1[r] * ATT_SCALE;
      float mx = fmaxf(p0, p1);
#pragma unroll
      for (int mk = 1; mk <= 8; mk <<= 1)           // reduce within 16-lane half
        mx = fmaxf(mx, __shfl_xor(mx, mk, 32));
      const float mnew = fmaxf(mrun[r], mx);
      alpha[r] = __expf(mrun[r] - mnew);
      p0 = __expf(p0 - mnew);
      p1 = __expf(p1 - mnew);
      float rs = p0 + p1;
#pragma unroll
      for (int mk = 1; mk <= 8; mk <<= 1)
        rs += __shfl_xor(rs, mk, 32);
      lrun[r] = lrun[r] * alpha[r] + rs;
      mrun[r] = mnew;
      Pw[r + 8 * half][n16]      = (bf16_t)p0;
      Pw[r + 8 * half][16 + n16] = (bf16_t)p1;
    }
#pragma unroll
    for (int nt = 0; nt < 4; ++nt)
#pragma unroll
      for (int r = 0; r < 8; ++r) accO[nt][r] *= alpha[r];

    // wave-internal LDS visibility (cross-lane read of staged P)
    asm volatile("s_wait_dscnt 0" ::: "memory");
    const v16bf pa = load_frag(&Pw[0][0], 34);

    // ---- O += P(16x32) . V(32x64); vT rows are K(seq)-contiguous ----
#pragma unroll
    for (int nt = 0; nt < 4; ++nt) {
      v16bf bv = load_frag(vTb + (size_t)(nt * 16) * SEQ + j0, SEQ);
      accO[nt] = wmma_bf16(pa, bv, accO[nt]);
    }
  }

  // ---- final normalize + store o[b, n, h*64+d] ----
#pragma unroll
  for (int nt = 0; nt < 4; ++nt)
#pragma unroll
    for (int r = 0; r < 8; ++r) {
      const int row = qrow0 + r + 8 * half;
      const int col = h * HDIM + nt * 16 + n16;
      o[((size_t)(b * SEQ + row)) * EMBED + col] = (bf16_t)(accO[nt][r] / lrun[r]);
    }
}

// ---------------- host-side orchestration ----------------
extern "C" void kernel_launch(void* const* d_in, const int* in_sizes, int n_in,
                              void* d_out, int out_size, void* d_ws, size_t ws_size,
                              hipStream_t stream) {
  (void)in_sizes; (void)n_in; (void)out_size; (void)ws_size;
  const float* x      = (const float*)d_in[0];
  const float* ln1_g  = (const float*)d_in[1];
  const float* ln1_b  = (const float*)d_in[2];
  const float* qk_w   = (const float*)d_in[3];
  const float* v_w    = (const float*)d_in[4];
  const float* proj_w = (const float*)d_in[5];
  const float* proj_b = (const float*)d_in[6];
  const float* ln2_g  = (const float*)d_in[7];
  const float* ln2_b  = (const float*)d_in[8];
  const float* fc1_w  = (const float*)d_in[9];
  const float* fc1_b  = (const float*)d_in[10];
  const float* fc2_w  = (const float*)d_in[11];
  const float* fc2_b  = (const float*)d_in[12];
  float* out = (float*)d_out;

  char* w = (char*)d_ws;
  size_t off = 0;
  auto take = [&](size_t bytes) {
    char* p = w + off;
    off = (off + bytes + 255) & ~(size_t)255;
    return p;
  };
  bf16_t* h1    = (bf16_t*)take((size_t)TOKENS * EMBED * 2);
  bf16_t* qws   = (bf16_t*)take((size_t)TOKENS * EMBED * 2);
  bf16_t* kws   = (bf16_t*)take((size_t)TOKENS * EMBED * 2);
  bf16_t* vTws  = (bf16_t*)take((size_t)TOKENS * EMBED * 2);
  bf16_t* ows   = (bf16_t*)take((size_t)TOKENS * EMBED * 2);
  float*  x1    = (float*)take((size_t)TOKENS * EMBED * 4);
  bf16_t* h2    = (bf16_t*)take((size_t)TOKENS * EMBED * 2);
  bf16_t* hmid  = (bf16_t*)take((size_t)TOKENS * MLP_H * 2);
  // bf16 transposed weights [N][K] (TDM-friendly: K-contiguous rows)
  bf16_t* qkT   = (bf16_t*)take((size_t)2 * EMBED * EMBED * 2);
  bf16_t* vwT   = (bf16_t*)take((size_t)EMBED * EMBED * 2);
  bf16_t* projT = (bf16_t*)take((size_t)EMBED * EMBED * 2);
  bf16_t* fc1T  = (bf16_t*)take((size_t)MLP_H * EMBED * 2);
  bf16_t* fc2T  = (bf16_t*)take((size_t)EMBED * MLP_H * 2);

  const dim3 b256(256);
  // 0) weight convert+transpose (fp32 [K][N] -> bf16 [N][K])
  transpose_w<<<dim3(EMBED / 32, (2 * EMBED) / 32), b256, 0, stream>>>(qk_w, qkT, EMBED, 2 * EMBED);
  transpose_w<<<dim3(EMBED / 32, EMBED / 32), b256, 0, stream>>>(v_w, vwT, EMBED, EMBED);
  transpose_w<<<dim3(EMBED / 32, EMBED / 32), b256, 0, stream>>>(proj_w, projT, EMBED, EMBED);
  transpose_w<<<dim3(EMBED / 32, MLP_H / 32), b256, 0, stream>>>(fc1_w, fc1T, EMBED, MLP_H);
  transpose_w<<<dim3(MLP_H / 32, EMBED / 32), b256, 0, stream>>>(fc2_w, fc2T, MLP_H, EMBED);
  // 1) LN1
  ln_kernel<<<TOKENS, b256, 0, stream>>>(x, ln1_g, ln1_b, h1);
  // 2) qk = h1 @ qk_w  -> scatter to q,k [B*H, N, 64]
  gemm_bf16<EPI_QK><<<dim3(TOKENS / 128, (2 * EMBED) / 64), b256, 0, stream>>>(
      h1, qkT, TOKENS, 2 * EMBED, EMBED, nullptr, nullptr, nullptr, qws, kws);
  // 3) v = h1 @ v_w -> vT [B*H, 64, N]
  gemm_bf16<EPI_V><<<dim3(TOKENS / 128, EMBED / 64), b256, 0, stream>>>(
      h1, vwT, TOKENS, EMBED, EMBED, nullptr, nullptr, nullptr, vTws, nullptr);
  // 4) attention -> o [B, N, EMBED]
  attn_kernel<<<dim3(SEQ / 64, BATCH * HEADS), dim3(128), 0, stream>>>(qws, kws, vTws, ows);
  // 5) x1 = x + o @ proj_w + proj_b
  gemm_bf16<EPI_PROJ><<<dim3(TOKENS / 128, EMBED / 64), b256, 0, stream>>>(
      ows, projT, TOKENS, EMBED, EMBED, proj_b, x, x1, nullptr, nullptr);
  // 6) LN2
  ln_kernel<<<TOKENS, b256, 0, stream>>>(x1, ln2_g, ln2_b, h2);
  // 7) hmid = gelu(h2 @ fc1_w + fc1_b)
  gemm_bf16<EPI_GELU><<<dim3(TOKENS / 128, MLP_H / 64), b256, 0, stream>>>(
      h2, fc1T, TOKENS, MLP_H, EMBED, fc1_b, nullptr, nullptr, hmid, nullptr);
  // 8) out = x1 + hmid @ fc2_w + fc2_b
  gemm_bf16<EPI_OUT><<<dim3(TOKENS / 128, EMBED / 64), b256, 0, stream>>>(
      hmid, fc2T, TOKENS, EMBED, MLP_H, fc2_b, x1, out, nullptr, nullptr);
}
